// GATGCN_5858335392234
// MI455X (gfx1250) — compile-verified, matrix operand order checked by reading, not run on hardware
//
#include <hip/hip_runtime.h>

typedef __attribute__((ext_vector_type(16))) _Float16 v16h;
typedef __attribute__((ext_vector_type(8)))  float    v8f;

#define HID 64
#define NSLOPE 0.2f

// ---------------- init: h1=0, out=b2 broadcast, per-node scalars ----------------
__global__ void gg_init(float* __restrict__ h1, float* __restrict__ out,
                        const float* __restrict__ b2,
                        float* __restrict__ emax, float* __restrict__ denom,
                        float* __restrict__ deg, int N) {
    int i = blockIdx.x * blockDim.x + threadIdx.x;
    if (i >= N * HID) return;
    int c = i & (HID - 1);
    h1[i] = 0.0f;
    out[i] = b2[c];
    int n = i >> 6;
    if (c == 0) {
        emax[n]  = -__builtin_inff();
        denom[n] = 0.0f;
        deg[n]   = 0.0f;
    }
}

// ------- convert weights f32 -> f16, pre-swizzled into WMMA B-fragment layout -----
// Fragment layout: frag[((kc*4 + ct)*32 + lane)*16 + e] where lane owns column
// n = ct*16 + (lane&15) and k = kc*32 + (lane>>4)*8 + (e<8 ? e : e+8).
__device__ __forceinline__ void cvt_frag(const float* __restrict__ W,
                                         _Float16* __restrict__ Wf, int j) {
    int e    = j & 15;
    int lane = (j >> 4) & 31;
    int ct   = (j >> 9) & 3;
    int kc   = j >> 11;
    int n = ct * 16 + (lane & 15);
    int k = kc * 32 + ((lane >> 4) << 3) + (e < 8 ? e : e + 8);
    Wf[j] = (_Float16)W[k * HID + n];
}

__global__ void gg_cvt_w(const float* __restrict__ W1, const float* __restrict__ W2,
                         _Float16* __restrict__ W1f, _Float16* __restrict__ W2f) {
    int i = blockIdx.x * blockDim.x + threadIdx.x;
    if (i < 128 * HID)                      cvt_frag(W1, W1f, i);
    else if (i < 128 * HID + HID * HID)     cvt_frag(W2, W2f, i - 128 * HID);
}

// ---------------- WMMA GEMM: C[M x 64] = A[M x K](f32) * Bf(f16 fragments) --------
// One wave computes one 16x16 tile of C. Requires M % 16 == 0.
template <int K>
__global__ void gg_gemm_wmma(const float* __restrict__ A, const _Float16* __restrict__ Bf,
                             float* __restrict__ C, int M) {
    int wave = (blockIdx.x * blockDim.x + threadIdx.x) >> 5;
    int lane = threadIdx.x & 31;
    int rtiles = M >> 4;
    int rt = wave >> 2;          // 64/16 = 4 column tiles
    int ct = wave & 3;
    if (rt >= rtiles) return;

    int l15   = lane & 15;
    int hi    = lane >> 4;
    int khalf = hi * 8;          // K sub-offset per ISA 16-bit A layout
    int row   = rt * 16 + l15;
    int nCol  = ct * 16 + l15;

    const float* arow = A + (size_t)row * K;
    v8f acc = {};

    #pragma unroll
    for (int kc = 0; kc < K; kc += 32) {
        int k0 = kc + khalf;
        const float4* p = (const float4*)(arow + k0);
        float4 f0 = p[0], f1 = p[1];
        const float4* q = (const float4*)(arow + k0 + 16);
        float4 f2 = q[0], f3 = q[1];
        v16h a;
        a[0]=(_Float16)f0.x;  a[1]=(_Float16)f0.y;  a[2]=(_Float16)f0.z;  a[3]=(_Float16)f0.w;
        a[4]=(_Float16)f1.x;  a[5]=(_Float16)f1.y;  a[6]=(_Float16)f1.z;  a[7]=(_Float16)f1.w;
        a[8]=(_Float16)f2.x;  a[9]=(_Float16)f2.y;  a[10]=(_Float16)f2.z; a[11]=(_Float16)f2.w;
        a[12]=(_Float16)f3.x; a[13]=(_Float16)f3.y; a[14]=(_Float16)f3.z; a[15]=(_Float16)f3.w;
        // pre-swizzled B fragment: one contiguous 32-byte load per lane
        v16h b = *(const v16h*)(Bf + ((size_t)(((kc >> 5) << 2) + ct) * 32 + lane) * 16);
        acc = __builtin_amdgcn_wmma_f32_16x16x32_f16(
            /*neg_a=*/false, a, /*neg_b=*/false, b,
            /*c_mod=*/(short)0, acc, /*reuse_a=*/false, /*reuse_b=*/false);
    }

    // D layout: VGPR i -> M = rt*16 + khalf + i, N = ct*16 + (lane&15)
    float* crow = C + (size_t)(rt * 16 + khalf) * HID + nCol;
    #pragma unroll
    for (int i = 0; i < 8; ++i) crow[(size_t)i * HID] = acc[i];
}

// ---------------- per-node attention scores: a_src/a_dst = h . att ----------------
__global__ void gg_att_scores(const float* __restrict__ h,
                              const float* __restrict__ att_s, const float* __restrict__ att_d,
                              float* __restrict__ a_s, float* __restrict__ a_d, int N) {
    int wave = (blockIdx.x * blockDim.x + threadIdx.x) >> 5;
    int lane = threadIdx.x & 31;
    if (wave >= N) return;
    const float* hr = h + (size_t)wave * HID;
    float x0 = hr[lane], x1 = hr[lane + 32];
    float s = x0 * att_s[lane] + x1 * att_s[lane + 32];
    float d = x0 * att_d[lane] + x1 * att_d[lane + 32];
    #pragma unroll
    for (int off = 16; off > 0; off >>= 1) {
        s += __shfl_down(s, off, 32);
        d += __shfl_down(d, off, 32);
    }
    if (lane == 0) { a_s[wave] = s; a_d[wave] = d; }
}

__device__ __forceinline__ void atomicMaxF(float* p, float v) {
    if (v >= 0.0f) atomicMax((int*)p, __float_as_int(v));
    else           atomicMin((unsigned int*)p, __float_as_uint(v));
}

// ---------------- edge pass 1: e = leaky_relu(a_src[s]+a_dst[d]); segment max ------
__global__ void gg_edge_max(const int* __restrict__ ei, int E, int EA,
                            const float* __restrict__ a_s, const float* __restrict__ a_d,
                            float* __restrict__ e_buf, float* __restrict__ emax) {
    int idx = blockIdx.x * blockDim.x + threadIdx.x;
    if (idx >= EA) return;
    int s, d;
    if (idx < E) { s = ei[idx]; d = ei[E + idx]; }
    else         { s = d = idx - E; }
    float e = a_s[s] + a_d[d];
    e = (e > 0.0f) ? e : NSLOPE * e;
    e_buf[idx] = e;
    atomicMaxF(&emax[d], e);
}

// ---------------- edge pass 2: exp, denom, degree ----------------
__global__ void gg_edge_exp(const int* __restrict__ ei, int E, int EA,
                            const float* __restrict__ emax,
                            float* __restrict__ e_buf, float* __restrict__ denom,
                            float* __restrict__ deg) {
    int idx = blockIdx.x * blockDim.x + threadIdx.x;
    if (idx >= EA) return;
    int d = (idx < E) ? ei[E + idx] : (idx - E);
    float ex = __expf(e_buf[idx] - emax[d]);
    e_buf[idx] = ex;
    atomicAdd(&denom[d], ex);
    atomicAdd(&deg[d], 1.0f);
}

// ---------------- edge pass 3: h1[dst] += alpha * h[src], wave per edge ------------
__global__ void gg_gat_scatter(const int* __restrict__ ei, int E, int EA,
                               const float* __restrict__ e_buf, const float* __restrict__ denom,
                               const float* __restrict__ h, float* __restrict__ h1) {
    // wave id is uniform across the wave: force scalar so index/score loads go SMEM
    int wave = __builtin_amdgcn_readfirstlane((blockIdx.x * blockDim.x + threadIdx.x) >> 5);
    int lane = threadIdx.x & 31;
    if (wave >= EA) return;
    int s, d;
    if (wave < E) { s = ei[wave]; d = ei[E + wave]; }
    else          { s = d = wave - E; }
    float alpha = e_buf[wave] / denom[d];
    const float* hs = h + (size_t)s * HID;
    float* hd = h1 + (size_t)d * HID;
    atomicAdd(&hd[lane],      alpha * hs[lane]);
    atomicAdd(&hd[lane + 32], alpha * hs[lane + 32]);
}

// ---------------- bias + relu ----------------
__global__ void gg_bias_relu(float* __restrict__ h1, const float* __restrict__ b1, int N) {
    int i = blockIdx.x * blockDim.x + threadIdx.x;
    if (i >= N * HID) return;
    float v = h1[i] + b1[i & (HID - 1)];
    h1[i] = v > 0.0f ? v : 0.0f;
}

// ---------------- edge pass 4: GCN scatter with symmetric norm ----------------
__global__ void gg_gcn_scatter(const int* __restrict__ ei, int E, int EA,
                               const float* __restrict__ deg,
                               const float* __restrict__ h2, float* __restrict__ out) {
    int wave = __builtin_amdgcn_readfirstlane((blockIdx.x * blockDim.x + threadIdx.x) >> 5);
    int lane = threadIdx.x & 31;
    if (wave >= EA) return;
    int s, d;
    if (wave < E) { s = ei[wave]; d = ei[E + wave]; }
    else          { s = d = wave - E; }
    float norm = rsqrtf(fmaxf(deg[s], 1.0f)) * rsqrtf(fmaxf(deg[d], 1.0f));
    const float* hs = h2 + (size_t)s * HID;
    float* od = out + (size_t)d * HID;
    atomicAdd(&od[lane],      norm * hs[lane]);
    atomicAdd(&od[lane + 32], norm * hs[lane + 32]);
}

extern "C" void kernel_launch(void* const* d_in, const int* in_sizes, int n_in,
                              void* d_out, int out_size, void* d_ws, size_t ws_size,
                              hipStream_t stream) {
    const float* x       = (const float*)d_in[0];
    const int*   ei      = (const int*)  d_in[1];
    const float* W1      = (const float*)d_in[2];
    const float* att_src = (const float*)d_in[3];
    const float* att_dst = (const float*)d_in[4];
    const float* b1      = (const float*)d_in[5];
    const float* W2      = (const float*)d_in[6];
    const float* b2      = (const float*)d_in[7];
    float* out = (float*)d_out;

    const int N  = in_sizes[0] / 128;
    const int E  = in_sizes[1] / 2;
    const int EA = E + N;

    // workspace carve-out (floats)
    float* h     = (float*)d_ws;           // N*64
    float* h1    = h     + (size_t)N * HID;
    float* h2    = h1    + (size_t)N * HID;
    float* a_s   = h2    + (size_t)N * HID;
    float* a_d   = a_s   + N;
    float* emax  = a_d   + N;
    float* denom = emax  + N;
    float* deg   = denom + N;
    float* e_buf = deg   + N;              // EA
    _Float16* W1f = (_Float16*)(e_buf + EA);
    _Float16* W2f = W1f + 128 * HID;

    const int T = 256;
    gg_init<<<((size_t)N * HID + T - 1) / T, T, 0, stream>>>(h1, out, b2, emax, denom, deg, N);
    gg_cvt_w<<<(128 * HID + HID * HID + T - 1) / T, T, 0, stream>>>(W1, W2, W1f, W2f);
    // GEMM1: h = x @ W1   (K = 128)
    {
        int waves  = ((N + 15) / 16) * 4;
        int blocks = (waves * 32 + T - 1) / T;
        gg_gemm_wmma<128><<<blocks, T, 0, stream>>>(x, W1f, h, N);
    }
    gg_att_scores<<<(N + 7) / 8, T, 0, stream>>>(h, att_src, att_dst, a_s, a_d, N);
    gg_edge_max<<<(EA + T - 1) / T, T, 0, stream>>>(ei, E, EA, a_s, a_d, e_buf, emax);
    gg_edge_exp<<<(EA + T - 1) / T, T, 0, stream>>>(ei, E, EA, emax, e_buf, denom, deg);
    gg_gat_scatter<<<(EA + 7) / 8, T, 0, stream>>>(ei, E, EA, e_buf, denom, h, h1);
    gg_bias_relu<<<((size_t)N * HID + T - 1) / T, T, 0, stream>>>(h1, b1, N);
    // GEMM2: h2 = h1 @ W2  (K = 64)
    {
        int waves  = ((N + 15) / 16) * 4;
        int blocks = (waves * 32 + T - 1) / T;
        gg_gemm_wmma<64><<<blocks, T, 0, stream>>>(h1, W2f, h2, N);
    }
    gg_gcn_scatter<<<(EA + 7) / 8, T, 0, stream>>>(ei, E, EA, deg, h2, out);
}
